// VectorQuantize_17222818857311
// MI455X (gfx1250) — matrix operand order repaired and checked by reading, not compiled, not run on hardware
//
#include <hip/hip_runtime.h>

#define N_PTS 65536
#define CDIM  64
#define KCB   4096
#define TK    128   // codebook entries staged per chunk
#define PITCH 72    // f16 elements per LDS row (144B: conflict-free row starts)
#define NCHUNK (KCB / TK)

typedef __attribute__((ext_vector_type(16))) _Float16 v16h;
typedef __attribute__((ext_vector_type(8)))  _Float16 v8h;
typedef __attribute__((ext_vector_type(8)))  float    v8f;

// ---------------------------------------------------------------------------
// Prep: cb16[k][c] = f16(codebook), c2[k] = ||codebook[k]||^2
// one wave32 per codebook row
// ---------------------------------------------------------------------------
__global__ __launch_bounds__(256) void vq_prep(const float* __restrict__ cb,
                                               _Float16* __restrict__ cb16,
                                               float* __restrict__ c2) {
    const int wave = threadIdx.x >> 5;
    const int lane = threadIdx.x & 31;
    const int row  = blockIdx.x * 8 + wave;
    const float2 v = *(const float2*)(cb + (size_t)row * CDIM + lane * 2);
    union { _Float16 h[2]; unsigned u; } pk;
    pk.h[0] = (_Float16)v.x;
    pk.h[1] = (_Float16)v.y;
    ((unsigned*)cb16)[row * 32 + lane] = pk.u;
    float s = v.x * v.x + v.y * v.y;
    s += __shfl_xor(s, 1, 32);
    s += __shfl_xor(s, 2, 32);
    s += __shfl_xor(s, 4, 32);
    s += __shfl_xor(s, 8, 32);
    s += __shfl_xor(s, 16, 32);
    if (lane == 0) c2[row] = s;
}

// ---------------------------------------------------------------------------
// Main: block = 8 waves x 32 rows = 256 points.
// Each wave holds two 16-row A tiles (x scaled by -2, f16) in registers and
// sweeps all 4096 codebook entries with f16 WMMA / f32 accum. Codebook chunks
// are DMA'd into ping-pong LDS with global_load_async_to_lds_b128 (ASYNCcnt),
// overlapping the copy of chunk ch+1 with the WMMA work of chunk ch.
// Argmin: column index packed into low 12 mantissa bits -> v_min_f32 chain.
// ---------------------------------------------------------------------------
__global__ __launch_bounds__(256) void vq_main(const float* __restrict__ x,
                                               const float* __restrict__ cb,
                                               const _Float16* __restrict__ cb16,
                                               const float* __restrict__ c2g,
                                               float* __restrict__ outq,
                                               int*   __restrict__ outi,
                                               float* __restrict__ outd) {
    __shared__ __align__(16) _Float16 sB[2][TK * PITCH];
    __shared__ float sC2[2][TK];
    __shared__ int   sIdx[8 * 32];

    const int tid  = threadIdx.x;
    const int wave = tid >> 5;
    const int lane = tid & 31;
    const int n16  = lane & 15;
    const int half = lane >> 4;
    const int rowBase = blockIdx.x * 256 + wave * 32;   // 32 rows per wave

    // ---- Load two A tiles (rows 0..15 and 16..31), scaled by -2, f16 ------
    // f16 A 16x32 layout: half0 -> K {0..7,16..23}, half1 -> K {8..15,24..31}
    const int kb = half * 8;
    v16h a0, a1, a2, a3;
    float x2t[2];
#pragma unroll
    for (int t = 0; t < 2; ++t) {
        const float* xr = x + (size_t)(rowBase + t * 16 + n16) * CDIM;
        float part = 0.f;
        v16h lo, hi;
#pragma unroll
        for (int e = 0; e < 8; ++e) {
            float v0 = xr[kb + e];      part += v0 * v0; lo[e]     = (_Float16)(-2.f * v0);
            float v1 = xr[16 + kb + e]; part += v1 * v1; lo[8 + e] = (_Float16)(-2.f * v1);
            float v2 = xr[32 + kb + e]; part += v2 * v2; hi[e]     = (_Float16)(-2.f * v2);
            float v3 = xr[48 + kb + e]; part += v3 * v3; hi[8 + e] = (_Float16)(-2.f * v3);
        }
        x2t[t] = part + __shfl_xor(part, 16, 32);
        if (t == 0) { a0 = lo; a1 = hi; } else { a2 = lo; a3 = hi; }
    }

    float minv[16];
#pragma unroll
    for (int i = 0; i < 16; ++i) minv[i] = 3.0e38f;

    const unsigned ldsBase[2] = { (unsigned)(uintptr_t)&sB[0][0],
                                  (unsigned)(uintptr_t)&sB[1][0] };

    // ---- async stage of one 128-entry f16 chunk into LDS buffer `buf` -----
    auto stage = [&](int ch, int buf) {
        const _Float16* src = cb16 + (size_t)ch * TK * CDIM;
#pragma unroll
        for (int i = 0; i < 4; ++i) {
            int task = i * 256 + tid;        // 1024 tasks = 128 rows x 8 segs
            int r    = task >> 3;
            int seg  = task & 7;
            unsigned lds = ldsBase[buf] + (unsigned)(r * (PITCH * 2) + seg * 16);
            unsigned long long ga =
                (unsigned long long)(uintptr_t)(src + r * CDIM + seg * 8);
            asm volatile("global_load_async_to_lds_b128 %0, %1, off"
                         :: "v"(lds), "v"(ga) : "memory");
        }
        if (tid < TK) sC2[buf][tid] = c2g[ch * TK + tid];
    };

    stage(0, 0);
    asm volatile("s_wait_asynccnt 0x0" ::: "memory");
    __syncthreads();

    for (int ch = 0; ch < NCHUNK; ++ch) {
        const int buf = ch & 1;
        if (ch + 1 < NCHUNK) stage(ch + 1, buf ^ 1);   // DMA overlaps compute

#pragma unroll
        for (int g = 0; g < TK / 16; ++g) {
            // B 32x16 layout: half0 -> K {0..7,16..23}, half1 -> K {8..15,24..31}
            const _Float16* br = &sB[buf][(g * 16 + n16) * PITCH + half * 8];
            v8h l0 = *(const v8h*)(br);
            v8h l1 = *(const v8h*)(br + 16);
            v8h l2 = *(const v8h*)(br + 32);
            v8h l3 = *(const v8h*)(br + 48);
            v16h b0 = __builtin_shufflevector(l0, l1, 0,1,2,3,4,5,6,7,8,9,10,11,12,13,14,15);
            v16h b1 = __builtin_shufflevector(l2, l3, 0,1,2,3,4,5,6,7,8,9,10,11,12,13,14,15);

            const float    cc  = sC2[buf][g * 16 + n16];
            const unsigned col = (unsigned)(ch * TK + g * 16 + n16);

            v8f acc = {};
            acc = __builtin_amdgcn_wmma_f32_16x16x32_f16(false, a0, false, b0,
                                                         (short)0, acc, false, false);
            acc = __builtin_amdgcn_wmma_f32_16x16x32_f16(false, a1, false, b1,
                                                         (short)0, acc, false, false);
#pragma unroll
            for (int i = 0; i < 8; ++i) {
                float t = acc[i] + cc;  // c2 - 2 x.c
                unsigned u = (__float_as_uint(t) & 0xFFFFF000u) | col;
                minv[i] = fminf(minv[i], __uint_as_float(u));
            }

            v8f acc2 = {};
            acc2 = __builtin_amdgcn_wmma_f32_16x16x32_f16(false, a2, false, b0,
                                                          (short)0, acc2, false, false);
            acc2 = __builtin_amdgcn_wmma_f32_16x16x32_f16(false, a3, false, b1,
                                                          (short)0, acc2, false, false);
#pragma unroll
            for (int i = 0; i < 8; ++i) {
                float t = acc2[i] + cc;
                unsigned u = (__float_as_uint(t) & 0xFFFFF000u) | col;
                minv[8 + i] = fminf(minv[8 + i], __uint_as_float(u));
            }
        }

        asm volatile("s_wait_asynccnt 0x0" ::: "memory");
        __syncthreads();
    }

    // ---- cross-lane argmin: butterfly within each 16-lane half ------------
#pragma unroll
    for (int i = 0; i < 16; ++i) {
        float v = minv[i];
        v = fminf(v, __shfl_xor(v, 1, 16));
        v = fminf(v, __shfl_xor(v, 2, 16));
        v = fminf(v, __shfl_xor(v, 4, 16));
        v = fminf(v, __shfl_xor(v, 8, 16));
        minv[i] = v;
    }

    // slot t*8+i: lanes 0-15 -> row t*16+i, lanes 16-31 -> row t*16+8+i.
#pragma unroll
    for (int t = 0; t < 2; ++t) {
        int outRow = -1, slot = 0;
        if (lane < 8)        { slot = t * 8 + lane;        outRow = t * 16 + lane; }
        else if (lane >= 24) { slot = t * 8 + (lane - 24); outRow = t * 16 + 8 + (lane - 24); }
        if (outRow >= 0) {
            float enc = minv[0];
#pragma unroll
            for (int i = 1; i < 16; ++i) enc = (slot == i) ? minv[i] : enc;
            int   idx  = (int)(__float_as_uint(enc) & 0xFFFu);
            float dist = sqrtf(fmaxf(x2t[t] + enc, 0.f));
            outi[rowBase + outRow] = idx;
            outd[rowBase + outRow] = dist;
            sIdx[wave * 32 + outRow] = idx;
        }
    }
    __syncthreads();

    // ---- gather quantized = codebook[idx] (wave copies its 32 rows) -------
    for (int r = 0; r < 32; ++r) {
        int ci = sIdx[wave * 32 + r];
        float2 v = *(const float2*)(cb + (size_t)ci * CDIM + lane * 2);
        *(float2*)(outq + (size_t)(rowBase + r) * CDIM + lane * 2) = v;
    }
}

// ---------------------------------------------------------------------------
extern "C" void kernel_launch(void* const* d_in, const int* in_sizes, int n_in,
                              void* d_out, int out_size, void* d_ws, size_t ws_size,
                              hipStream_t stream) {
    (void)in_sizes; (void)n_in; (void)out_size; (void)ws_size;
    const float* x  = (const float*)d_in[0];
    const float* cb = (const float*)d_in[1];
    _Float16* cb16 = (_Float16*)d_ws;                              // 512 KB
    float*    c2   = (float*)((char*)d_ws + (size_t)KCB * CDIM * 2); // 16 KB
    float* outq = (float*)d_out;                                   // [N, C]
    int*   outi = (int*)(outq + (size_t)N_PTS * CDIM);             // [N] int32
    float* outd = (float*)(outq + (size_t)N_PTS * CDIM + N_PTS);   // [N]

    vq_prep<<<KCB / 8, 256, 0, stream>>>(cb, cb16, c2);
    vq_main<<<N_PTS / 256, 256, 0, stream>>>(x, cb, cb16, c2, outq, outi, outd);
}
